// RealTimeMotionGenerator_70463233458825
// MI455X (gfx1250) — matrix operand order repaired and checked by reading, not compiled, not run on hardware
//
#include <hip/hip_runtime.h>
#include <hip/hip_bf16.h>
#include <math.h>

// ---------------- problem constants ----------------
#define TEXT_DIM 256
#define POSE_DIM 72
#define POSE_PAD 96      // K padded to 3x32 for WMMA
#define HDIM     256
#define GDIM     768     // 3*H
#define IN_DIM   328
#define MLPDIM   256
#define OUT_PAD  80      // MLP2 N padded to 5x16
#define MROWS    32      // batch rows per workgroup (2 M-tiles of 16)

typedef unsigned short u16;
typedef __attribute__((ext_vector_type(16))) __bf16 v16bf;
typedef __attribute__((ext_vector_type(8)))  __bf16 v8bf;
typedef __attribute__((ext_vector_type(8)))  float  v8f;

// f32 -> bf16 bits, round-to-nearest-even
static __device__ __forceinline__ u16 f2bf(float f) {
    union { float f; unsigned u; } x; x.f = f;
    unsigned u = x.u;
    unsigned r = u + 0x7FFFu + ((u >> 16) & 1u);
    return (u16)(r >> 16);
}

// A-fragment (16x32 bf16) per ISA layout:
// lane m = lane&15 is the row; lane[4] selects K-halves:
//   sel=0: K0..7 and K16..23 ; sel=1: K8..15 and K24..31
static __device__ __forceinline__ v16bf load_a(const u16* base, int pitch, int lane) {
    int m   = lane & 15;
    int sel = (lane >> 4) & 1;
    const u16* p = base + m * pitch + sel * 8;
    v8bf lo = *(const v8bf*)(p);
    v8bf hi = *(const v8bf*)(p + 16);
    return __builtin_shufflevector(lo, hi, 0,1,2,3,4,5,6,7,8,9,10,11,12,13,14,15);
}

// B-fragment (32x16 bf16): weights stored transposed [K][N] row-major;
// lane n = lane&15 is the column; lane[4] selects K0..7,K16..23 or K8..15,K24..31.
static __device__ __forceinline__ v16bf load_b(const u16* base_k0_n0, int pitch, int lane) {
    const int n = lane & 15, sel = (lane >> 4) & 1;
    const u16* p = base_k0_n0 + n + (size_t)(sel * 8) * pitch;
    v16bf r;
#pragma unroll
    for (int s = 0; s < 8; ++s) {
        r[s]     = __builtin_bit_cast(__bf16, p[(size_t)s * pitch]);
        r[8 + s] = __builtin_bit_cast(__bf16, p[(size_t)(16 + s) * pitch]);
    }
    return r;
}

static __device__ __forceinline__ v8f wmma_bf16(v16bf a, v16bf b, v8f c) {
    return __builtin_amdgcn_wmma_f32_16x16x32_bf16(
        /*neg_a=*/false, a, /*neg_b=*/false, b,
        /*c_mod=*/(short)0, c, /*reuse_a=*/false, /*reuse_b=*/false);
}

static __device__ __forceinline__ float sigmoidf_(float x) {
    return 1.0f / (1.0f + expf(-x));
}

// ---------------- prep 1: weight transpose/convert to bf16 ----------------
__global__ void prep_weights_kernel(const float* __restrict__ W_ih,
                                    const float* __restrict__ W_hh,
                                    const float* __restrict__ W1f,
                                    const float* __restrict__ W2f,
                                    const float* __restrict__ text,
                                    u16* __restrict__ whh_t, u16* __restrict__ wiht_t,
                                    u16* __restrict__ wihp_t, u16* __restrict__ w1_t,
                                    u16* __restrict__ w2_t, u16* __restrict__ text_bf,
                                    long n_text) {
    long i = (long)blockIdx.x * blockDim.x + threadIdx.x;
    const long N0 = (long)HDIM * GDIM;          // whh_t  [256][768]
    const long N1 = (long)TEXT_DIM * GDIM;      // wiht_t [256][768]
    const long N2 = (long)POSE_PAD * GDIM;      // wihp_t [96][768]
    const long N3 = (long)HDIM * MLPDIM;        // w1_t   [256][256]
    const long N4 = (long)MLPDIM * OUT_PAD;     // w2_t   [256][80]
    if (i < N0) { long k = i / GDIM, n = i % GDIM;
        whh_t[i] = f2bf(W_hh[n * HDIM + k]); return; }
    i -= N0;
    if (i < N1) { long k = i / GDIM, n = i % GDIM;
        wiht_t[i] = f2bf(W_ih[n * IN_DIM + k]); return; }
    i -= N1;
    if (i < N2) { long k = i / GDIM, n = i % GDIM;
        wihp_t[i] = (k < POSE_DIM) ? f2bf(W_ih[n * IN_DIM + TEXT_DIM + k]) : (u16)0; return; }
    i -= N2;
    if (i < N3) { long k = i / MLPDIM, n = i % MLPDIM;
        w1_t[i] = f2bf(W1f[n * HDIM + k]); return; }
    i -= N3;
    if (i < N4) { long k = i / OUT_PAD, n = i % OUT_PAD;
        w2_t[i] = (n < POSE_DIM) ? f2bf(W2f[n * MLPDIM + k]) : (u16)0; return; }
    i -= N4;
    if (i < n_text) { text_bf[i] = f2bf(text[i]); return; }
}

// ---------------- prep 2: gi_text = text @ W_ih_text^T + b_ih ----------------
__global__ __launch_bounds__(256, 4)
void gitext_kernel(const u16* __restrict__ text_bf, const u16* __restrict__ wiht_t,
                   const float* __restrict__ b_ih, float* __restrict__ gitext) {
    const int tid = threadIdx.x, lane = tid & 31, wave = tid >> 5;
    const int gw = blockIdx.x * 8 + wave;          // one wave per 16x16 tile
    const int mt = gw / 48, nt = gw - mt * 48;
    const int b0 = mt << 4, n0 = nt << 4;
    const u16* abase = text_bf + (size_t)b0 * TEXT_DIM;
    v8f acc = {};
#pragma unroll
    for (int kc = 0; kc < 8; ++kc) {
        v16bf a = load_a(abase + kc * 32, TEXT_DIM, lane);
        v16bf b = load_b(wiht_t + ((size_t)kc * 32) * GDIM + n0, GDIM, lane);
        acc = wmma_bf16(a, b, acc);
    }
    const int n = lane & 15, mb = (lane >> 4) << 3;
    const float bias = b_ih[n0 + n];
#pragma unroll
    for (int j = 0; j < 8; ++j)
        gitext[(size_t)(b0 + mb + j) * GDIM + n0 + n] = acc[j] + bias;
}

// ---------------- main persistent GRU + MLP kernel ----------------
// 256 blocks x 256 threads (8 waves), 32 batch rows per block (2 M-tiles),
// each weight B-fragment is loaded once and used for both M-tiles.
// ~208KB dynamic LDS (gfx1250: 320KB/WGP).
__global__ __launch_bounds__(256, 1)
void gru_kernel(const float* __restrict__ hidden, const float* __restrict__ prev,
                const int* __restrict__ steps_p,
                const float* __restrict__ b_hh, const float* __restrict__ b1,
                const float* __restrict__ b2,
                const u16* __restrict__ whh_t, const u16* __restrict__ wihp_t,
                const u16* __restrict__ w1_t, const u16* __restrict__ w2_t,
                const float* __restrict__ gitext,
                float* __restrict__ out_poses, float* __restrict__ out_h) {
    extern __shared__ char smem_raw[];
    float* grz   = (float*)smem_raw;           // [32][512] i+h sums for r,z gates
    float* gin   = grz  + MROWS * 512;         // [32][256] i_n
    float* ghn   = gin  + MROWS * 256;         // [32][256] h_n (+b_hh)
    float* hcur  = ghn  + MROWS * 256;         // [32][256] f32 hidden
    float* poseS = hcur + MROWS * 256;         // [32][80]  f32 pose
    u16*   h_bf  = (u16*)(poseS + MROWS * 80); // [32][256] bf16 hidden
    u16*   x_bf  = h_bf + MROWS * 256;         // [32][96]  bf16 pose (K-padded)
    u16*   a1s   = x_bf + MROWS * 96;          // [32][256] bf16 relu(MLP1)

    const int tid = threadIdx.x, lane = tid & 31, wave = tid >> 5;
    const int b0 = blockIdx.x * MROWS;
    const int steps = steps_p[0];
    const int n_lane = lane & 15;
    const int mb = (lane >> 4) << 3;

    // init hidden + pose tiles
    for (int idx = tid; idx < MROWS * 256; idx += 256) {
        int m = idx >> 8, j = idx & 255;
        float h = hidden[(size_t)(b0 + m) * HDIM + j];
        hcur[idx] = h;
        h_bf[idx] = f2bf(h);
    }
    for (int idx = tid; idx < MROWS * 96; idx += 256) {
        int m = idx / 96, c = idx - m * 96;
        float p = (c < POSE_DIM) ? prev[(size_t)(b0 + m) * POSE_DIM + c] : 0.0f;
        x_bf[idx] = f2bf(p);
        if (c < POSE_DIM) poseS[m * 80 + c] = p;
    }

    for (int t = 0; t < steps; ++t) {
        __syncthreads();   // h_bf / x_bf ready

        // ---- gates: gh = h@Whh^T , gi_pose = x@Wihp^T (kept separate for n-gate)
        {
            v16bf ah0[8], ah1[8];
#pragma unroll
            for (int kc = 0; kc < 8; ++kc) {
                ah0[kc] = load_a(h_bf + kc * 32, 256, lane);
                ah1[kc] = load_a(h_bf + 16 * 256 + kc * 32, 256, lane);
            }
            v16bf ax0[3], ax1[3];
#pragma unroll
            for (int kc = 0; kc < 3; ++kc) {
                ax0[kc] = load_a(x_bf + kc * 32, 96, lane);
                ax1[kc] = load_a(x_bf + 16 * 96 + kc * 32, 96, lane);
            }
#pragma unroll
            for (int i = 0; i < 6; ++i) {
                const int nt = wave * 6 + i;       // 48 N-tiles over 8 waves
                const int n0 = nt << 4;
                v8f h0 = {}, h1 = {}, i0 = {}, i1 = {};
#pragma unroll
                for (int kc = 0; kc < 8; ++kc) {
                    v16bf b = load_b(whh_t + ((size_t)kc * 32) * GDIM + n0, GDIM, lane);
                    h0 = wmma_bf16(ah0[kc], b, h0);
                    h1 = wmma_bf16(ah1[kc], b, h1);
                }
#pragma unroll
                for (int kc = 0; kc < 3; ++kc) {
                    v16bf b = load_b(wihp_t + ((size_t)kc * 32) * GDIM + n0, GDIM, lane);
                    i0 = wmma_bf16(ax0[kc], b, i0);
                    i1 = wmma_bf16(ax1[kc], b, i1);
                }
                const int col = n0 + n_lane;
                const float bhh = b_hh[col];
                if (nt < 32) {                      // r,z gates: sum i+h parts
#pragma unroll
                    for (int j = 0; j < 8; ++j) {
                        int m0 = mb + j, m1 = 16 + mb + j;
                        grz[m0 * 512 + col] = h0[j] + i0[j]
                            + gitext[(size_t)(b0 + m0) * GDIM + col] + bhh;
                        grz[m1 * 512 + col] = h1[j] + i1[j]
                            + gitext[(size_t)(b0 + m1) * GDIM + col] + bhh;
                    }
                } else {                            // n gate: keep i_n, h_n separate
                    const int c = col - 512;
#pragma unroll
                    for (int j = 0; j < 8; ++j) {
                        int m0 = mb + j, m1 = 16 + mb + j;
                        gin[m0 * 256 + c] = i0[j] + gitext[(size_t)(b0 + m0) * GDIM + col];
                        ghn[m0 * 256 + c] = h0[j] + bhh;
                        gin[m1 * 256 + c] = i1[j] + gitext[(size_t)(b0 + m1) * GDIM + col];
                        ghn[m1 * 256 + c] = h1[j] + bhh;
                    }
                }
            }
        }
        __syncthreads();

        // ---- elementwise GRU update
        for (int idx = tid; idx < MROWS * 256; idx += 256) {
            int m = idx >> 8, j = idx & 255;
            float r  = sigmoidf_(grz[m * 512 + j]);
            float z  = sigmoidf_(grz[m * 512 + 256 + j]);
            float nn = tanhf(gin[idx] + r * ghn[idx]);
            float h  = hcur[idx];
            float hn = (1.0f - z) * nn + z * h;
            hcur[idx] = hn;
            h_bf[idx] = f2bf(hn);
        }
        __syncthreads();

        // ---- MLP1: a1 = relu(h@W1^T + b1)
        {
            v16bf am0[8], am1[8];
#pragma unroll
            for (int kc = 0; kc < 8; ++kc) {
                am0[kc] = load_a(h_bf + kc * 32, 256, lane);
                am1[kc] = load_a(h_bf + 16 * 256 + kc * 32, 256, lane);
            }
#pragma unroll
            for (int i = 0; i < 2; ++i) {
                const int n0 = (wave * 2 + i) << 4;
                v8f a0 = {}, a1v = {};
#pragma unroll
                for (int kc = 0; kc < 8; ++kc) {
                    v16bf b = load_b(w1_t + ((size_t)kc * 32) * MLPDIM + n0, MLPDIM, lane);
                    a0  = wmma_bf16(am0[kc], b, a0);
                    a1v = wmma_bf16(am1[kc], b, a1v);
                }
                const float bb = b1[n0 + n_lane];
#pragma unroll
                for (int j = 0; j < 8; ++j) {
                    a1s[(mb + j) * 256 + n0 + n_lane]      = f2bf(fmaxf(a0[j]  + bb, 0.0f));
                    a1s[(16 + mb + j) * 256 + n0 + n_lane] = f2bf(fmaxf(a1v[j] + bb, 0.0f));
                }
            }
        }
        __syncthreads();

        // ---- MLP2: delta = a1@W2^T + b2 ; pose += delta ; emit poses[b,t,:]
        if (wave < 5) {                         // 80 padded cols = 5 N-tiles
            v16bf aa0[8], aa1[8];
#pragma unroll
            for (int kc = 0; kc < 8; ++kc) {
                aa0[kc] = load_a(a1s + kc * 32, 256, lane);
                aa1[kc] = load_a(a1s + 16 * 256 + kc * 32, 256, lane);
            }
            const int n0 = wave << 4;
            v8f d0 = {}, d1 = {};
#pragma unroll
            for (int kc = 0; kc < 8; ++kc) {
                v16bf b = load_b(w2_t + ((size_t)kc * 32) * OUT_PAD + n0, OUT_PAD, lane);
                d0 = wmma_bf16(aa0[kc], b, d0);
                d1 = wmma_bf16(aa1[kc], b, d1);
            }
            const int col = n0 + n_lane;
            if (col < POSE_DIM) {
                const float bb = b2[col];
#pragma unroll
                for (int j = 0; j < 8; ++j) {
                    int m0 = mb + j, m1 = 16 + mb + j;
                    float np0 = poseS[m0 * 80 + col] + d0[j] + bb;
                    float np1 = poseS[m1 * 80 + col] + d1[j] + bb;
                    poseS[m0 * 80 + col] = np0;
                    poseS[m1 * 80 + col] = np1;
                    x_bf[m0 * 96 + col] = f2bf(np0);
                    x_bf[m1 * 96 + col] = f2bf(np1);
                    out_poses[((size_t)(b0 + m0) * steps + t) * POSE_DIM + col] = np0;
                    out_poses[((size_t)(b0 + m1) * steps + t) * POSE_DIM + col] = np1;
                }
            }
        }
    }

    __syncthreads();
    for (int idx = tid; idx < MROWS * 256; idx += 256) {
        int m = idx >> 8, j = idx & 255;
        out_h[(size_t)(b0 + m) * HDIM + j] = hcur[idx];
    }
}

// ---------------- host side ----------------
extern "C" void kernel_launch(void* const* d_in, const int* in_sizes, int n_in,
                              void* d_out, int out_size, void* d_ws, size_t ws_size,
                              hipStream_t stream) {
    const float* text   = (const float*)d_in[0];
    const float* prev   = (const float*)d_in[1];
    const float* hidden = (const float*)d_in[2];
    const int*   steps  = (const int*)d_in[3];
    const float* W_ih   = (const float*)d_in[4];
    const float* W_hh   = (const float*)d_in[5];
    const float* b_ih   = (const float*)d_in[6];
    const float* b_hh   = (const float*)d_in[7];
    const float* W1     = (const float*)d_in[8];
    const float* b1     = (const float*)d_in[9];
    const float* W2     = (const float*)d_in[10];
    const float* b2     = (const float*)d_in[11];

    const int Bsz = in_sizes[0] / TEXT_DIM;    // 8192

    // workspace carve (bf16 transposed weights + gi_text)
    char* ws = (char*)d_ws;
    size_t off = 0;
    auto take = [&](size_t bytes) { size_t r = off; off += (bytes + 511) & ~(size_t)511; return r; };
    u16* whh_t   = (u16*)(ws + take((size_t)HDIM * GDIM * 2));
    u16* wiht_t  = (u16*)(ws + take((size_t)TEXT_DIM * GDIM * 2));
    u16* wihp_t  = (u16*)(ws + take((size_t)POSE_PAD * GDIM * 2));
    u16* w1_t    = (u16*)(ws + take((size_t)HDIM * MLPDIM * 2));
    u16* w2_t    = (u16*)(ws + take((size_t)MLPDIM * OUT_PAD * 2));
    u16* text_bf = (u16*)(ws + take((size_t)Bsz * TEXT_DIM * 2));
    float* gitext = (float*)(ws + take((size_t)Bsz * GDIM * 4));

    // prep 1: convert/transpose weights + text to bf16
    {
        long n_text = (long)Bsz * TEXT_DIM;
        long total = (long)HDIM * GDIM + (long)TEXT_DIM * GDIM + (long)POSE_PAD * GDIM
                   + (long)HDIM * MLPDIM + (long)MLPDIM * OUT_PAD + n_text;
        int grid = (int)((total + 255) / 256);
        prep_weights_kernel<<<grid, 256, 0, stream>>>(
            W_ih, W_hh, W1, W2, text,
            whh_t, wiht_t, wihp_t, w1_t, w2_t, text_bf, n_text);
    }

    // prep 2: step-invariant gi_text (one 8192x768x256 GEMM via WMMA)
    {
        int tiles = (Bsz / 16) * (GDIM / 16);  // 24576
        gitext_kernel<<<tiles / 8, 256, 0, stream>>>(text_bf, wiht_t, b_ih, gitext);
    }

    // main GRU: 32 batch rows per block, steps sequential, ~208KB dynamic LDS
    {
        const int SMEM = (MROWS * 512 + 3 * MROWS * 256 + MROWS * 80) * 4
                       + (MROWS * 256 + MROWS * 96 + MROWS * 256) * 2;   // 212992 B
        (void)hipFuncSetAttribute((const void*)gru_kernel,
                                  hipFuncAttributeMaxDynamicSharedMemorySize, SMEM);
        float* out_poses = (float*)d_out;
        float* out_h     = (float*)d_out + ((size_t)out_size - (size_t)Bsz * HDIM);
        gru_kernel<<<Bsz / MROWS, 256, SMEM, stream>>>(
            hidden, prev, steps, b_hh, b1, b2,
            whh_t, wihp_t, w1_t, w2_t, gitext, out_poses, out_h);
    }
}
